// CrossAttention_26130581029556
// MI455X (gfx1250) — matrix operand order repaired and checked by reading, not compiled, not run on hardware
//
#include <hip/hip_runtime.h>
#include <math.h>

// ---------------- types for WMMA ----------------
typedef __attribute__((ext_vector_type(16))) __bf16   v16bf;
typedef __attribute__((ext_vector_type(8)))  float    v8f;
typedef __attribute__((ext_vector_type(4)))  unsigned u32x4;

union Frag { v16bf v; u32x4 q[2]; };
static_assert(sizeof(Frag) == 32, "frag size");

__device__ __forceinline__ unsigned short f2bf(float f) {
    unsigned u = __float_as_uint(f);
    u += 0x7FFFu + ((u >> 16) & 1u);          // round-to-nearest-even
    return (unsigned short)(u >> 16);
}
__device__ __forceinline__ float bf2f(unsigned short h) {
    return __uint_as_float(((unsigned)h) << 16);
}

// ---------------- constants ----------------
constexpr int GN = 16384;   // N = H*W
constexpr int GK = 384;     // K = C
constexpr int CH = 384;
constexpr int QKVCH = 1152; // 3*C
constexpr long QKV_ELEMS = (long)4 * QKVCH * GN;

// =====================================================================
// bf16x3 WMMA GEMM:  C[b] = A[M,384] * B[b][384,16384]
// block tile 128x128, BK=32; 8 waves of 32; wave tile 64x32 (4x2 wmma tiles)
// =====================================================================
__global__ __launch_bounds__(256) void gemm_bf16x3(
    const float* __restrict__ A, const float* __restrict__ B,
    float* __restrict__ C, long bStrideB, long bStrideC)
{
    const int t     = threadIdx.x;
    const int lane  = t & 31;
    const int wave  = t >> 5;
    const int waveM = wave >> 2;     // 0..1
    const int waveN = wave & 3;      // 0..3
    const int laneM = lane & 15;
    const int half  = lane >> 4;

    const long batchB = (long)blockIdx.z * bStrideB;
    const long batchC = (long)blockIdx.z * bStrideC;
    const int m0 = blockIdx.y * 128;
    const int n0 = blockIdx.x * 128;

    // pitch 40 bf16 = 80B: 16B aligned rows, conflict-friendly
    __shared__ unsigned short Ahi[128 * 40];
    __shared__ unsigned short Alo[128 * 40];
    __shared__ unsigned short Bhi[128 * 40];  // stored transposed: [n][k]
    __shared__ unsigned short Blo[128 * 40];

    v8f acc[4][2] = {};

    for (int kc = 0; kc < GK; kc += 32) {
        __syncthreads();
#pragma unroll
        for (int r = 0; r < 16; ++r) {
            int idx = r * 256 + t;
            {   // A tile: 128(M) x 32(K), coalesced along K
                int row = idx >> 5, k = idx & 31;
                float v = A[(long)(m0 + row) * GK + kc + k];
                unsigned short h = f2bf(v);
                Ahi[row * 40 + k] = h;
                Alo[row * 40 + k] = f2bf(v - bf2f(h));
            }
            {   // B tile: 32(K) x 128(N), coalesced along N, stored [n][k]
                int k = idx >> 7, n = idx & 127;
                float v = B[batchB + (long)(kc + k) * GN + n0 + n];
                unsigned short h = f2bf(v);
                Bhi[n * 40 + k] = h;
                Blo[n * 40 + k] = f2bf(v - bf2f(h));
            }
        }
        __syncthreads();

        Frag ah[4], al[4], bh[2], bl[2];
#pragma unroll
        for (int i = 0; i < 4; ++i) {
            int row = waveM * 64 + i * 16 + laneM;
            const u32x4* ph = (const u32x4*)(&Ahi[row * 40]);
            const u32x4* pl = (const u32x4*)(&Alo[row * 40]);
            // A 16-bit layout: elems 0..7 -> K = 8*half+0..7 ; elems 8..15 -> K = 16+8*half+0..7
            ah[i].q[0] = ph[half];     ah[i].q[1] = ph[2 + half];
            al[i].q[0] = pl[half];     al[i].q[1] = pl[2 + half];
        }
#pragma unroll
        for (int j = 0; j < 2; ++j) {
            int n = waveN * 32 + j * 16 + laneM;
            const u32x4* ph = (const u32x4*)(&Bhi[n * 40]);
            const u32x4* pl = (const u32x4*)(&Blo[n * 40]);
            // B 16-bit layout: lane = N, elems 0..7 -> K = 8*half+0..7 ; elems 8..15 -> K = 16+8*half+0..7
            bh[j].q[0] = ph[half]; bh[j].q[1] = ph[2 + half];
            bl[j].q[0] = pl[half]; bl[j].q[1] = pl[2 + half];
        }
#pragma unroll
        for (int i = 0; i < 4; ++i)
#pragma unroll
            for (int j = 0; j < 2; ++j) {
                acc[i][j] = __builtin_amdgcn_wmma_f32_16x16x32_bf16(
                    false, ah[i].v, false, bh[j].v, (short)0, acc[i][j], false, false);
                acc[i][j] = __builtin_amdgcn_wmma_f32_16x16x32_bf16(
                    false, ah[i].v, false, bl[j].v, (short)0, acc[i][j], false, false);
                acc[i][j] = __builtin_amdgcn_wmma_f32_16x16x32_bf16(
                    false, al[i].v, false, bh[j].v, (short)0, acc[i][j], false, false);
            }
    }

    // C/D layout: VGPR r -> M = r + 8*half, N = laneM
#pragma unroll
    for (int i = 0; i < 4; ++i) {
        int mt = m0 + waveM * 64 + i * 16;
#pragma unroll
        for (int j = 0; j < 2; ++j) {
            int nt = n0 + waveN * 32 + j * 16 + laneM;
#pragma unroll
            for (int r = 0; r < 8; ++r) {
                int mm = mt + r + 8 * half;
                C[batchC + (long)mm * GN + nt] = acc[i][j][r];
            }
        }
    }
}

// =====================================================================
// depthwise 3x3, SAME zero padding, cross-correlation (XLA conv)
// in/out: [4,1152,128,128], W: [1152,1,3,3]
// =====================================================================
__global__ __launch_bounds__(256) void dw_kernel(
    const float* __restrict__ in, const float* __restrict__ Wdw,
    float* __restrict__ out)
{
    long gid = (long)blockIdx.x * 256 + threadIdx.x;
    int  n   = (int)(gid & 16383);
    long bc  = gid >> 14;
    int  ch  = (int)(bc % QKVCH);
    int  hh  = n >> 7, ww = n & 127;
    const float* base = in + (bc << 14);
    const float* w = Wdw + ch * 9;
    float s = 0.f;
#pragma unroll
    for (int dy = 0; dy < 3; ++dy)
#pragma unroll
        for (int dx = 0; dx < 3; ++dx) {
            int y = hh + dy - 1, x = ww + dx - 1;
            if ((unsigned)y < 128u && (unsigned)x < 128u)
                s += w[dy * 3 + dx] * base[y * 128 + x];
        }
    out[gid] = s;
}

// =====================================================================
// row l2-norm scales: scales[id] = 1/max(||row||, 1e-12)
// id in [0,3072): which = id/1536 (0=q rows, 1=k rows), b, c
// =====================================================================
__global__ __launch_bounds__(256) void rownorm_kernel(
    const float* __restrict__ qkv, float* __restrict__ scales)
{
    int id = blockIdx.x;
    int which = id / 1536, rem = id % 1536;
    int b = rem / CH, c = rem % CH;
    const float* row = qkv + ((long)b * QKVCH + which * CH + c) * GN;
    int t = threadIdx.x;
    float s = 0.f;
#pragma unroll 4
    for (int i = 0; i < 64; ++i) {
        float v = row[t + i * 256];
        s += v * v;
    }
    __shared__ float red[256];
    red[t] = s;
    __syncthreads();
    for (int off = 128; off > 0; off >>= 1) {
        if (t < off) red[t] += red[t + off];
        __syncthreads();
    }
    if (t == 0) scales[id] = 1.0f / fmaxf(sqrtf(red[0]), 1e-12f);
}

// =====================================================================
// attention logits + softmax per (b,h): S = softmax(scale .* (q k^T) * temp)
// q rows: ch h*48..+47 ; k rows: 384+h*48..+47 ; K = 16384 chunked by 128
// =====================================================================
__global__ __launch_bounds__(256) void attn_kernel(
    const float* __restrict__ qkv, const float* __restrict__ scales,
    const float* __restrict__ temp, float* __restrict__ attnW)
{
    int b = blockIdx.x >> 3, h = blockIdx.x & 7;
    const float* qb = qkv + ((long)b * QKVCH + h * 48) * GN;
    const float* kb = qkv + ((long)b * QKVCH + CH + h * 48) * GN;

    __shared__ float qs[48 * 132];
    __shared__ float ks[48 * 132];
    __shared__ float S[48 * 48];

    int t = threadIdx.x;
    int tc = t >> 4, td = t & 15;     // 16x16 thread grid, 3x3 register tile
    float acc[3][3] = {};

    for (int k0 = 0; k0 < GN; k0 += 128) {
        __syncthreads();
#pragma unroll
        for (int r = 0; r < 24; ++r) {
            int idx = r * 256 + t;
            int row = idx >> 7, kk = idx & 127;
            qs[row * 132 + kk] = qb[(long)row * GN + k0 + kk];
            ks[row * 132 + kk] = kb[(long)row * GN + k0 + kk];
        }
        __syncthreads();
        for (int kk = 0; kk < 128; ++kk) {
            float qv[3], kv[3];
#pragma unroll
            for (int i = 0; i < 3; ++i) qv[i] = qs[(tc + 16 * i) * 132 + kk];
#pragma unroll
            for (int j = 0; j < 3; ++j) kv[j] = ks[(td + 16 * j) * 132 + kk];
#pragma unroll
            for (int i = 0; i < 3; ++i)
#pragma unroll
                for (int j = 0; j < 3; ++j) acc[i][j] += qv[i] * kv[j];
        }
    }

    float tpr = temp[h];
#pragma unroll
    for (int i = 0; i < 3; ++i)
#pragma unroll
        for (int j = 0; j < 3; ++j) {
            int c = tc + 16 * i, d = td + 16 * j;
            float sc = scales[b * CH + h * 48 + c] *
                       scales[1536 + b * CH + h * 48 + d] * tpr;
            S[c * 48 + d] = acc[i][j] * sc;
        }
    __syncthreads();
    if (t < 48) {
        float m = -INFINITY;
        for (int d = 0; d < 48; ++d) m = fmaxf(m, S[t * 48 + d]);
        float s = 0.f;
        for (int d = 0; d < 48; ++d) {
            float e = expf(S[t * 48 + d] - m);
            S[t * 48 + d] = e;
            s += e;
        }
        float inv = 1.0f / s;
        for (int d = 0; d < 48; ++d) S[t * 48 + d] *= inv;
    }
    __syncthreads();
#pragma unroll
    for (int p = 0; p < 9; ++p) {
        int pi = t + p * 256;
        attnW[(long)blockIdx.x * 2304 + pi] = S[pi];
    }
}

// =====================================================================
// out[b][h*48+c][n] = sum_d attn[c,d] * v[d,n] ; v rows = 768 + h*48 + d
// grid (64 n-chunks, 8 heads, 4 batches), 256 threads: one n per thread
// =====================================================================
__global__ __launch_bounds__(256) void av_kernel(
    const float* __restrict__ qkv, const float* __restrict__ attnW,
    float* __restrict__ outb)
{
    int b = blockIdx.z, h = blockIdx.y;
    int t = threadIdx.x;
    long n = (long)blockIdx.x * 256 + t;

    __shared__ float At[2304];
#pragma unroll
    for (int p = 0; p < 9; ++p)
        At[t + p * 256] = attnW[(long)(b * 8 + h) * 2304 + t + p * 256];
    __syncthreads();

    const float* vb = qkv + ((long)b * QKVCH + 2 * CH + h * 48) * GN;
    float vreg[48];
#pragma unroll
    for (int d = 0; d < 48; ++d) vreg[d] = vb[(long)d * GN + n];

    float* ob = outb + ((long)b * CH + h * 48) * GN;
#pragma unroll 4
    for (int c = 0; c < 48; ++c) {
        float s = 0.f;
#pragma unroll
        for (int d = 0; d < 48; ++d) s += At[c * 48 + d] * vreg[d];
        ob[(long)c * GN + n] = s;
    }
}

// =====================================================================
extern "C" void kernel_launch(void* const* d_in, const int* in_sizes, int n_in,
                              void* d_out, int out_size, void* d_ws, size_t ws_size,
                              hipStream_t stream) {
    const float* x    = (const float*)d_in[0];
    const float* y    = (const float*)d_in[1];
    const float* Wq   = (const float*)d_in[2];
    const float* Wkv  = (const float*)d_in[3];
    const float* Wdw  = (const float*)d_in[4];
    const float* Wproj= (const float*)d_in[5];
    const float* temp = (const float*)d_in[6];
    float* out = (float*)d_out;

    float* buf0  = (float*)d_ws;            // qkv pre-dw, later attn-out (compact)
    float* buf1  = buf0 + QKV_ELEMS;        // qkv post-dw
    float* scal  = buf1 + QKV_ELEMS;        // 3072 floats
    float* attnW = scal + 3072;             // 4*8*48*48 floats

    const long xBS   = (long)CH * GN;       // 384*16384
    const long qkvBS = (long)QKVCH * GN;    // 1152*16384

    // q = Wq * x  -> buf0 rows [0,384)
    gemm_bf16x3<<<dim3(128, 3, 4), 256, 0, stream>>>(Wq, x, buf0, xBS, qkvBS);
    // kv = Wkv * y -> buf0 rows [384,1152)
    gemm_bf16x3<<<dim3(128, 6, 4), 256, 0, stream>>>(Wkv, y, buf0 + xBS, xBS, qkvBS);
    // depthwise 3x3
    dw_kernel<<<294912, 256, 0, stream>>>(buf0, Wdw, buf1);
    // l2-norm scales for q and k rows
    rownorm_kernel<<<3072, 256, 0, stream>>>(buf1, scal);
    // attention weights (48x48 per (b,h)) with softmax
    attn_kernel<<<32, 256, 0, stream>>>(buf1, scal, temp, attnW);
    // attn @ v -> buf0 (compact [4,384,16384])
    av_kernel<<<dim3(64, 8, 4), 256, 0, stream>>>(buf1, attnW, buf0);
    // projection: out = Wproj * buf0
    gemm_bf16x3<<<dim3(128, 3, 4), 256, 0, stream>>>(Wproj, buf0, out, xBS, xBS);
}
